// Multidecoder_Attention_81295140978725
// MI455X (gfx1250) — matrix operand order repaired and checked by reading, not compiled, not run on hardware
//
#include <hip/hip_runtime.h>

// ---------------------------------------------------------------------------
// Multidecoder window attention for MI455X (gfx1250, wave32, WMMA bf16,
// async global->LDS staging, double-buffered LDS tiles).
//   per window w (sequential, color carried):
//     1) k_qkv_gemm : X(4552x768)f32 x WqkvT(bf16) -> Q/K/V (bf16, V transposed)
//     2) k_attn     : 96 (b,h) flash attention, T=569, d=64 (pure bf16 WMMA)
//     3) k_proj_gemm: A(4552x768)bf16 x WprojT(bf16) -> patches / color (f32)
// ---------------------------------------------------------------------------

#define DIMC     768
#define HEADS    12
#define HEAD_DIM 64
#define NWIN     16
#define BSZ      8
#define NPATCH   256
#define NCLS     313
#define TTOK     569                 // T = 256 + 313
#define MTOT     (BSZ * TTOK)        // 4552
#define NQKV     (3 * DIMC)          // 2304
#define QK_SCALE 0.125f              // 64^-0.5
#define VSTRIDE  576                 // padded key stride for transposed V

typedef __bf16 bf16_t;
typedef bf16_t v16bf __attribute__((ext_vector_type(16)));
typedef float  v8f   __attribute__((ext_vector_type(8)));

// ---- gfx1250 async global->LDS copy (ASYNCcnt), with safe fallback --------
#if defined(__has_builtin)
#if __has_builtin(__builtin_amdgcn_global_load_async_to_lds_b128) && \
    __has_builtin(__builtin_amdgcn_s_wait_asynccnt)
#define HAVE_ASYNC_LDS 1
#endif
#endif

// builtin signature is (v4i addrspace(1)*, v4i addrspace(3)*, Ii, Ii) where
// v4i is a *generic* vector of 4 ints (vector_size, not ext_vector_type).
typedef int v4i_t __attribute__((vector_size(4 * sizeof(int))));

__device__ __forceinline__ void cp16_async(const void* g, void* l) {
#if defined(HAVE_ASYNC_LDS)
  __builtin_amdgcn_global_load_async_to_lds_b128(
      (__attribute__((address_space(1))) v4i_t*)const_cast<void*>(g),
      (__attribute__((address_space(3))) v4i_t*)l, 0, 0);
#else
  *(uint4*)l = *(const uint4*)g;
#endif
}
__device__ __forceinline__ void cp_commit_wait() {
#if defined(HAVE_ASYNC_LDS)
  __builtin_amdgcn_s_wait_asynccnt(0);
#endif
}

union Frag16 { v16bf v; uint4 q[2]; };

// A-fragment (16x32 bf16): lane r=lane&15, half=lane>>4.
// elems 0..15 = K(half*16 + 0..15) contig.
__device__ __forceinline__ v16bf ld_frag_a(const bf16_t* row, int half) {
  Frag16 f;
  f.q[0] = *(const uint4*)(row + half * 16);
  f.q[1] = *(const uint4*)(row + half * 16 + 8);
  return f.v;
}
// B-fragment (32x16 bf16): lane n=lane&15: elems = K(khalf*16 + 0..15) contig.
__device__ __forceinline__ v16bf ld_frag_b(const bf16_t* p) {
  Frag16 f;
  f.q[0] = *(const uint4*)(p);
  f.q[1] = *(const uint4*)(p + 8);
  return f.v;
}
__device__ __forceinline__ void st8_bf16(bf16_t* dst, const float* x) {
  union { bf16_t h[8]; uint4 q; } pk;
  #pragma unroll
  for (int i = 0; i < 8; ++i) pk.h[i] = (bf16_t)x[i];
  *(uint4*)dst = pk.q;
}

// ---------------- one-time helpers -----------------------------------------
__global__ void k_transpose_bf16(const float* __restrict__ w,
                                 bf16_t* __restrict__ wt, int K, int N) {
  int idx = blockIdx.x * blockDim.x + threadIdx.x;
  if (idx >= K * N) return;
  int n = idx % N, k = idx / N;
  wt[(size_t)n * K + k] = (bf16_t)w[idx];
}
__global__ void k_zero_bf16(bf16_t* __restrict__ p, int n) {
  int i = blockIdx.x * blockDim.x + threadIdx.x;
  if (i < n) p[i] = (bf16_t)0.f;
}

#define LDA 40
#define LDB 40

// ---------------- QKV GEMM: (4552 x 768)f32 x (768 x 2304)bf16 -------------
__global__ __launch_bounds__(256)
void k_qkv_gemm(const float* __restrict__ patch, const float* __restrict__ colorIn,
                const bf16_t* __restrict__ Bt, const float* __restrict__ bias,
                bf16_t* __restrict__ qB, bf16_t* __restrict__ kB,
                bf16_t* __restrict__ vB, int w) {
  __shared__ bf16_t As[2][128][LDA];
  __shared__ bf16_t Bs[2][128][LDB];
  const int tid = threadIdx.x, lane = tid & 31, wave = tid >> 5;
  const int half = lane >> 4, nlo = lane & 15;
  const int wm = wave >> 1, wn = wave & 1;
  const int blkm = blockIdx.x, blkn = blockIdx.y;

  v8f zero8 = {0.f, 0.f, 0.f, 0.f, 0.f, 0.f, 0.f, 0.f};
  v8f acc[2][4];
  #pragma unroll
  for (int mt = 0; mt < 2; ++mt)
    #pragma unroll
    for (int nt = 0; nt < 4; ++nt) acc[mt][nt] = zero8;

  const int arow = tid >> 1, acg = (tid & 1) * 16;
  const int gm_l = blkm * 128 + arow;
  const float* aSrc = nullptr;
  if (gm_l < MTOT) {
    int b = gm_l / TTOK, t = gm_l % TTOK;
    aSrc = (t < NPATCH)
         ? patch + ((size_t)((b * NWIN + w) * NPATCH + t)) * DIMC
         : colorIn + ((size_t)(b * NCLS + (t - NPATCH))) * DIMC;
  }
  const bf16_t* bSrc = Bt + (size_t)(blkn * 128 + arow) * DIMC;

  float tmp[16];
  auto loadA = [&](int kb) {
    if (aSrc) {
      const float* s = aSrc + kb + acg;
      #pragma unroll
      for (int i = 0; i < 16; i += 4) {
        float4 v4 = *(const float4*)(s + i);
        tmp[i] = v4.x; tmp[i + 1] = v4.y; tmp[i + 2] = v4.z; tmp[i + 3] = v4.w;
      }
    } else {
      #pragma unroll
      for (int i = 0; i < 16; ++i) tmp[i] = 0.f;
    }
  };
  auto stageB = [&](int bufi, int kb) {
    cp16_async(bSrc + kb + acg, &Bs[bufi][arow][acg]);
    cp16_async(bSrc + kb + acg + 8, &Bs[bufi][arow][acg + 8]);
  };

  // prologue: fill buffer 0
  stageB(0, 0);
  loadA(0);
  st8_bf16(&As[0][arow][acg], tmp);
  st8_bf16(&As[0][arow][acg + 8], tmp + 8);
  cp_commit_wait();
  __syncthreads();

  int buf = 0;
  for (int kb = 0; kb < DIMC; kb += 32) {
    const int nxt = kb + 32;
    if (nxt < DIMC) { stageB(buf ^ 1, nxt); loadA(nxt); }

    v16bf af[2], bfv[4];
    #pragma unroll
    for (int mt = 0; mt < 2; ++mt)
      af[mt] = ld_frag_a(&As[buf][wm * 32 + mt * 16 + nlo][0], half);
    #pragma unroll
    for (int nt = 0; nt < 4; ++nt)
      bfv[nt] = ld_frag_b(&Bs[buf][wn * 64 + nt * 16 + nlo][half * 16]);
    #pragma unroll
    for (int mt = 0; mt < 2; ++mt)
      #pragma unroll
      for (int nt = 0; nt < 4; ++nt)
        acc[mt][nt] = __builtin_amdgcn_wmma_f32_16x16x32_bf16(
            false, af[mt], false, bfv[nt], (short)0, acc[mt][nt], false, false);

    if (nxt < DIMC) {
      st8_bf16(&As[buf ^ 1][arow][acg], tmp);
      st8_bf16(&As[buf ^ 1][arow][acg + 8], tmp + 8);
      cp_commit_wait();
    }
    __syncthreads();
    buf ^= 1;
  }

  // epilogue: bf16 scatter; Q/K row-major per (b,h); V transposed [d][t]
  #pragma unroll
  for (int mt = 0; mt < 2; ++mt) {
    #pragma unroll
    for (int v = 0; v < 8; ++v) {
      int gm = blkm * 128 + wm * 32 + mt * 16 + v + half * 8;
      if (gm >= MTOT) continue;
      int b = gm / TTOK, t = gm % TTOK;
      #pragma unroll
      for (int nt = 0; nt < 4; ++nt) {
        int gn = blkn * 128 + wn * 64 + nt * 16 + nlo;
        float val = acc[mt][nt][v] + bias[gn];
        int which = gn / DIMC, c = gn % DIMC;
        int h = c >> 6, d = c & 63;
        size_t bh = (size_t)(b * HEADS + h);
        if (which == 0)
          qB[(bh * TTOK + t) * HEAD_DIM + d] = (bf16_t)val;
        else if (which == 1)
          kB[(bh * TTOK + t) * HEAD_DIM + d] = (bf16_t)val;
        else
          vB[(bh * HEAD_DIM + d) * VSTRIDE + t] = (bf16_t)val;
      }
    }
  }
}

// ---------------- flash attention per (b,h), 128 q-rows per block ----------
__global__ __launch_bounds__(256)
void k_attn(const bf16_t* __restrict__ qB, const bf16_t* __restrict__ kB,
            const bf16_t* __restrict__ vB, const int* __restrict__ mask,
            bf16_t* __restrict__ attnB, int w) {
  __shared__ bf16_t Ks[2][32][72];    // [key][d]
  __shared__ bf16_t Vs[2][64][40];    // [d][key] (pre-transposed in global)
  __shared__ bf16_t Ps[8][16][40];    // per-wave P tile [q][key]
  const int tid = threadIdx.x, lane = tid & 31, wave = tid >> 5;
  const int half = lane >> 4, nlo = lane & 15;
  const int bh = blockIdx.x, b = bh / HEADS, h = bh % HEADS;
  const bf16_t* Qbh = qB + (size_t)bh * TTOK * HEAD_DIM;
  const bf16_t* Kbh = kB + (size_t)bh * TTOK * HEAD_DIM;
  const bf16_t* Vbh = vB + (size_t)bh * HEAD_DIM * VSTRIDE;
  const int* mbase = mask + (size_t)(b * NWIN + w) * TTOK * TTOK;
  const int q0 = blockIdx.y * 128 + wave * 16;

  // Q fragments: direct bf16 loads (scale folded into S later)
  const int qr = min(q0 + nlo, TTOK - 1);
  const bf16_t* Qrow = Qbh + (size_t)qr * HEAD_DIM;
  v16bf qf[2];
  #pragma unroll
  for (int c = 0; c < 2; ++c) {
    Frag16 f;
    f.q[0] = *(const uint4*)(Qrow + c * 32 + half * 16);
    f.q[1] = *(const uint4*)(Qrow + c * 32 + half * 16 + 8);
    qf[c] = f.v;
  }

  v8f zero8 = {0.f, 0.f, 0.f, 0.f, 0.f, 0.f, 0.f, 0.f};
  v8f o[4];
  #pragma unroll
  for (int nt = 0; nt < 4; ++nt) o[nt] = zero8;
  float rm[8], rs[8];
  #pragma unroll
  for (int v = 0; v < 8; ++v) { rm[v] = -1e30f; rs[v] = 0.f; }

  const int kr = tid >> 3, kcg = (tid & 7) * 8;  // K staging: 32 rows x 64d
  const int dr = tid >> 2, vcg = (tid & 3) * 8;  // V staging: 64 rows x 32k

  auto stage = [&](int bufi, int kt) {
    int key = min(kt + kr, TTOK - 1);
    cp16_async(Kbh + (size_t)key * HEAD_DIM + kcg, &Ks[bufi][kr][kcg]);
    cp16_async(Vbh + (size_t)dr * VSTRIDE + kt + vcg, &Vs[bufi][dr][vcg]);
  };
  stage(0, 0);
  cp_commit_wait();
  __syncthreads();

  int buf = 0;
  for (int kt = 0; kt < TTOK; kt += 32) {
    const int nxt = kt + 32;
    if (nxt < TTOK) stage(buf ^ 1, nxt);

    // S = Q * K^T for 2 key tiles of 16
    v8f s[2];
    #pragma unroll
    for (int tt = 0; tt < 2; ++tt) {
      v8f sv = zero8;
      v16bf bk0 = ld_frag_b(&Ks[buf][tt * 16 + nlo][half * 16]);
      v16bf bk1 = ld_frag_b(&Ks[buf][tt * 16 + nlo][32 + half * 16]);
      sv = __builtin_amdgcn_wmma_f32_16x16x32_bf16(false, qf[0], false, bk0,
                                                   (short)0, sv, false, false);
      sv = __builtin_amdgcn_wmma_f32_16x16x32_bf16(false, qf[1], false, bk1,
                                                   (short)0, sv, false, false);
      s[tt] = sv;
    }
    // scale + mask + key tail (C layout: row = v + 8*half, col = nlo)
    #pragma unroll
    for (int tt = 0; tt < 2; ++tt) {
      #pragma unroll
      for (int v = 0; v < 8; ++v) {
        int qrow = min(q0 + v + half * 8, TTOK - 1);
        int kcol = kt + tt * 16 + nlo;
        float sval = s[tt][v] * QK_SCALE;
        if (kcol >= TTOK || mbase[(size_t)qrow * TTOK + kcol] == 0)
          sval = -1e30f;
        s[tt][v] = sval;
      }
    }
    // online softmax: row reductions across the 16-lane half
    #pragma unroll
    for (int v = 0; v < 8; ++v) {
      float mx = fmaxf(s[0][v], s[1][v]);
      mx = fmaxf(mx, __shfl_xor(mx, 1, 32));
      mx = fmaxf(mx, __shfl_xor(mx, 2, 32));
      mx = fmaxf(mx, __shfl_xor(mx, 4, 32));
      mx = fmaxf(mx, __shfl_xor(mx, 8, 32));
      float mnew = fmaxf(rm[v], mx);
      float corr = __expf(rm[v] - mnew);
      float p0 = __expf(s[0][v] - mnew);
      float p1 = __expf(s[1][v] - mnew);
      float ps = p0 + p1;
      ps += __shfl_xor(ps, 1, 32);
      ps += __shfl_xor(ps, 2, 32);
      ps += __shfl_xor(ps, 4, 32);
      ps += __shfl_xor(ps, 8, 32);
      rs[v] = rs[v] * corr + ps;
      rm[v] = mnew;
      #pragma unroll
      for (int nt = 0; nt < 4; ++nt) o[nt][v] *= corr;
      Ps[wave][v + half * 8][nlo] = (bf16_t)p0;
      Ps[wave][v + half * 8][16 + nlo] = (bf16_t)p1;
    }
    // O += P * V
    v16bf pf = ld_frag_a(&Ps[wave][nlo][0], half);
    #pragma unroll
    for (int nt = 0; nt < 4; ++nt) {
      v16bf bv = ld_frag_b(&Vs[buf][nt * 16 + nlo][half * 16]);
      o[nt] = __builtin_amdgcn_wmma_f32_16x16x32_bf16(false, pf, false, bv,
                                                      (short)0, o[nt], false, false);
    }
    if (nxt < TTOK) cp_commit_wait();
    __syncthreads();
    buf ^= 1;
  }

  // write O (bf16) into (b, t, h*64+d) layout for the proj GEMM
  #pragma unroll
  for (int v = 0; v < 8; ++v) {
    int qrow = q0 + v + half * 8;
    if (qrow >= TTOK) continue;
    float inv = 1.0f / fmaxf(rs[v], 1e-30f);
    #pragma unroll
    for (int nt = 0; nt < 4; ++nt) {
      int d = nt * 16 + nlo;
      attnB[((size_t)(b * TTOK + qrow)) * DIMC + h * HEAD_DIM + d] =
          (bf16_t)(o[nt][v] * inv);
    }
  }
}

// ---------------- proj GEMM: (4552 x 768)bf16 x (768 x 768)bf16 ------------
__global__ __launch_bounds__(256)
void k_proj_gemm(const bf16_t* __restrict__ attnB, const bf16_t* __restrict__ Bt,
                 const float* __restrict__ bias, float* __restrict__ outPatch,
                 float* __restrict__ colorOut, int w) {
  __shared__ bf16_t As[2][128][LDA];
  __shared__ bf16_t Bs[2][128][LDB];
  const int tid = threadIdx.x, lane = tid & 31, wave = tid >> 5;
  const int half = lane >> 4, nlo = lane & 15;
  const int wm = wave >> 1, wn = wave & 1;
  const int blkm = blockIdx.x, blkn = blockIdx.y;

  v8f zero8 = {0.f, 0.f, 0.f, 0.f, 0.f, 0.f, 0.f, 0.f};
  v8f acc[2][4];
  #pragma unroll
  for (int mt = 0; mt < 2; ++mt)
    #pragma unroll
    for (int nt = 0; nt < 4; ++nt) acc[mt][nt] = zero8;

  const int arow = tid >> 1, acg = (tid & 1) * 16;
  const int gm_l = blkm * 128 + arow;
  const bf16_t* aSrc = (gm_l < MTOT) ? attnB + (size_t)gm_l * DIMC : nullptr;
  const bf16_t* bSrc = Bt + (size_t)(blkn * 128 + arow) * DIMC;

  auto stage = [&](int bufi, int kb) {
    if (aSrc) {
      cp16_async(aSrc + kb + acg, &As[bufi][arow][acg]);
      cp16_async(aSrc + kb + acg + 8, &As[bufi][arow][acg + 8]);
    } else {
      uint4 z = make_uint4(0u, 0u, 0u, 0u);
      *(uint4*)&As[bufi][arow][acg] = z;
      *(uint4*)&As[bufi][arow][acg + 8] = z;
    }
    cp16_async(bSrc + kb + acg, &Bs[bufi][arow][acg]);
    cp16_async(bSrc + kb + acg + 8, &Bs[bufi][arow][acg + 8]);
  };
  stage(0, 0);
  cp_commit_wait();
  __syncthreads();

  int buf = 0;
  for (int kb = 0; kb < DIMC; kb += 32) {
    const int nxt = kb + 32;
    if (nxt < DIMC) stage(buf ^ 1, nxt);

    v16bf af[2], bfv[4];
    #pragma unroll
    for (int mt = 0; mt < 2; ++mt)
      af[mt] = ld_frag_a(&As[buf][wm * 32 + mt * 16 + nlo][0], half);
    #pragma unroll
    for (int nt = 0; nt < 4; ++nt)
      bfv[nt] = ld_frag_b(&Bs[buf][wn * 64 + nt * 16 + nlo][half * 16]);
    #pragma unroll
    for (int mt = 0; mt < 2; ++mt)
      #pragma unroll
      for (int nt = 0; nt < 4; ++nt)
        acc[mt][nt] = __builtin_amdgcn_wmma_f32_16x16x32_bf16(
            false, af[mt], false, bfv[nt], (short)0, acc[mt][nt], false, false);

    if (nxt < DIMC) cp_commit_wait();
    __syncthreads();
    buf ^= 1;
  }

  #pragma unroll
  for (int mt = 0; mt < 2; ++mt) {
    #pragma unroll
    for (int v = 0; v < 8; ++v) {
      int gm = blkm * 128 + wm * 32 + mt * 16 + v + half * 8;
      if (gm >= MTOT) continue;
      int b = gm / TTOK, t = gm % TTOK;
      #pragma unroll
      for (int nt = 0; nt < 4; ++nt) {
        int gn = blkn * 128 + wn * 64 + nt * 16 + nlo;
        float val = acc[mt][nt][v] + bias[gn];
        if (t < NPATCH)
          outPatch[((size_t)((b * NWIN + w) * NPATCH + t)) * DIMC + gn] = val;
        else
          colorOut[((size_t)(b * NCLS + (t - NPATCH))) * DIMC + gn] = val;
      }
    }
  }
}

// ---------------------------------------------------------------------------
extern "C" void kernel_launch(void* const* d_in, const int* in_sizes, int n_in,
                              void* d_out, int out_size, void* d_ws,
                              size_t ws_size, hipStream_t stream) {
  const float* patch  = (const float*)d_in[0];
  const float* color0 = (const float*)d_in[1];
  const int*   mask   = (const int*)d_in[2];
  const float* qkv_w  = (const float*)d_in[3];
  const float* qkv_b  = (const float*)d_in[4];
  const float* proj_w = (const float*)d_in[5];
  const float* proj_b = (const float*)d_in[6];
  float* out = (float*)d_out;

  char* p = (char*)d_ws;
  bf16_t* wqkvT  = (bf16_t*)p; p += (size_t)NQKV * DIMC * sizeof(bf16_t);
  bf16_t* wprojT = (bf16_t*)p; p += (size_t)DIMC * DIMC * sizeof(bf16_t);
  bf16_t* qB     = (bf16_t*)p; p += (size_t)BSZ * HEADS * TTOK * HEAD_DIM * sizeof(bf16_t);
  bf16_t* kB     = (bf16_t*)p; p += (size_t)BSZ * HEADS * TTOK * HEAD_DIM * sizeof(bf16_t);
  bf16_t* vB     = (bf16_t*)p; p += (size_t)BSZ * HEADS * HEAD_DIM * VSTRIDE * sizeof(bf16_t);
  bf16_t* attnB  = (bf16_t*)p; p += (size_t)MTOT * DIMC * sizeof(bf16_t);
  float*  cb0    = (float*)p;  p += (size_t)BSZ * NCLS * DIMC * sizeof(float);
  float*  cb1    = (float*)p;

  k_transpose_bf16<<<(NQKV * DIMC + 255) / 256, 256, 0, stream>>>(qkv_w, wqkvT,
                                                                  DIMC, NQKV);
  k_transpose_bf16<<<(DIMC * DIMC + 255) / 256, 256, 0, stream>>>(proj_w, wprojT,
                                                                  DIMC, DIMC);
  { // zero V buffer once so its 7-column key padding can never be NaN
    int nv = BSZ * HEADS * HEAD_DIM * VSTRIDE;
    k_zero_bf16<<<(nv + 255) / 256, 256, 0, stream>>>(vB, nv);
  }

  const size_t patchOutElems = (size_t)BSZ * NWIN * NPATCH * DIMC;
  const float* colorIn = color0;
  for (int w = 0; w < NWIN; ++w) {
    float* colorOut = (w == NWIN - 1) ? (out + patchOutElems)
                                      : ((w & 1) ? cb1 : cb0);
    k_qkv_gemm<<<dim3(36, 18), 256, 0, stream>>>(patch, colorIn, wqkvT, qkv_b,
                                                 qB, kB, vB, w);
    k_attn<<<dim3(96, 5), 256, 0, stream>>>(qB, kB, vB, mask, attnB, w);
    k_proj_gemm<<<dim3(36, 6), 256, 0, stream>>>(attnB, wprojT, proj_b, out,
                                                 colorOut, w);
    colorIn = colorOut;
  }
}
